// MambaBlock_7335804142280
// MI455X (gfx1250) — matrix operand order repaired and checked by reading, not compiled, not run on hardware
//
#include <hip/hip_runtime.h>
#include <hip/hip_bf16.h>

// ---------------------------------------------------------------------------
// Mamba block for MI455X (gfx1250, wave32, WMMA, async global->LDS, double-
// buffered software pipeline, chunked parallel selective scan).
//   1) xr   = x @ W_in            (f16 WMMA GEMM, 4096x4096x1024)
//   2) xs   = silu(conv(xin)+b)   (elementwise; also emits xs16)
//   3) xdbl = xs @ W_x            (f16 WMMA GEMM, 4096x96x2048)
//   4) delta= softplus(xdbl[:, :64] @ W_dt + b_dt)
//   5) ys   = selective_scan      (chunked: passA locals -> combine -> passB)
//   6) out  = (ys * silu(res)) @ W_out
// ---------------------------------------------------------------------------

typedef _Float16 v16h __attribute__((ext_vector_type(16)));
typedef _Float16 h8   __attribute__((ext_vector_type(8)));
typedef float    v8f  __attribute__((ext_vector_type(8)));

#define BM 128
#define BN 128
#define BK 32
#define LDSS 40   // padded LDS row stride in halves (80B): spreads banks,
                  // every 16B chunk stays contiguous & 16B-aligned.

__device__ __forceinline__ uint32_t lds_addr_u32(const void* p) {
  // generic LDS pointers carry the LDS byte offset in the low 32 bits
  return (uint32_t)(uintptr_t)p;
}

__device__ __forceinline__ void async_b128(uint32_t lds_byte, const _Float16* g) {
  asm volatile("global_load_async_to_lds_b128 %0, %1, off"
               :: "v"(lds_byte), "v"(g) : "memory");
}

__device__ __forceinline__ void wait_async0() {
#if __has_builtin(__builtin_amdgcn_s_wait_asynccnt)
  __builtin_amdgcn_s_wait_asynccnt(0);
#else
  asm volatile("s_wait_asynccnt 0x0" ::: "memory");
#endif
}

// ----------------------------- WMMA GEMM -----------------------------------
// C[M,N] = A[M,K] @ B[K,N]; A f16 row-major [M][K], B f16 TRANSPOSED [N][K].
// f32 accumulate/output. M % 128 == 0, K % 32 == 0; N guarded per block.
// Double-buffered LDS, async copies for tile k+1 overlap WMMAs of tile k.
__global__ void __launch_bounds__(256)
gemm_wmma_f16(const _Float16* __restrict__ A, int lda,
              const _Float16* __restrict__ BT, int ldbt,
              float* __restrict__ C, int ldc,
              int M, int N, int K)
{
  __shared__ _Float16 As[2][BM * LDSS];
  __shared__ _Float16 Bs[2][BN * LDSS];

  const int tid  = threadIdx.x;
  const int lane = tid & 31;
  const int wave = tid >> 5;
  const int wm   = wave & 3;     // 4 waves along M, 32 rows each
  const int wn   = wave >> 2;    // 2 waves along N, 64 cols each
  const int l15  = lane & 15;
  const int hh   = lane >> 4;

  const int rowBase = blockIdx.y * BM;
  const int colBase = blockIdx.x * BN;
  const bool fullN  = (colBase + BN <= N);   // block-uniform guard

  const v8f vz = {0.f,0.f,0.f,0.f,0.f,0.f,0.f,0.f};
  v8f acc[2][4];
#pragma unroll
  for (int i = 0; i < 2; ++i)
#pragma unroll
    for (int j = 0; j < 4; ++j) acc[i][j] = vz;

  const h8 hz = {(_Float16)0,(_Float16)0,(_Float16)0,(_Float16)0,
                 (_Float16)0,(_Float16)0,(_Float16)0,(_Float16)0};

  // per-thread staging coordinates (512 x 16B chunks per operand tile)
  const int c0r  = (tid) >> 2,        c0k = (tid) & 3;
  const int c1r  = (tid + 256) >> 2,  c1k = (tid + 256) & 3;

  auto stage = [&](int k0, int buf) {
    // A tile: 128 rows x 4 chunks
    async_b128(lds_addr_u32(&As[buf][c0r * LDSS + c0k * 8]),
               A + (size_t)(rowBase + c0r) * lda + k0 + c0k * 8);
    async_b128(lds_addr_u32(&As[buf][c1r * LDSS + c1k * 8]),
               A + (size_t)(rowBase + c1r) * lda + k0 + c1k * 8);
    // B tile (transposed source): 128 rows x 4 chunks
    if (fullN) {
      async_b128(lds_addr_u32(&Bs[buf][c0r * LDSS + c0k * 8]),
                 BT + (size_t)(colBase + c0r) * ldbt + k0 + c0k * 8);
      async_b128(lds_addr_u32(&Bs[buf][c1r * LDSS + c1k * 8]),
                 BT + (size_t)(colBase + c1r) * ldbt + k0 + c1k * 8);
    } else {
      if (colBase + c0r < N)
        async_b128(lds_addr_u32(&Bs[buf][c0r * LDSS + c0k * 8]),
                   BT + (size_t)(colBase + c0r) * ldbt + k0 + c0k * 8);
      else
        *(h8*)&Bs[buf][c0r * LDSS + c0k * 8] = hz;
      if (colBase + c1r < N)
        async_b128(lds_addr_u32(&Bs[buf][c1r * LDSS + c1k * 8]),
                   BT + (size_t)(colBase + c1r) * ldbt + k0 + c1k * 8);
      else
        *(h8*)&Bs[buf][c1r * LDSS + c1k * 8] = hz;
    }
  };

  stage(0, 0);
  int buf = 0;
  for (int k0 = 0; k0 < K; k0 += BK, buf ^= 1) {
    wait_async0();        // own async copies of tile k done
    __syncthreads();      // -> every wave's copies done; prior reads retired
    if (k0 + BK < K) stage(k0 + BK, buf ^ 1);   // overlap with compute below

    v16h af[2], bf[4];
#pragma unroll
    for (int i = 0; i < 2; ++i) {
      int m = wm * 32 + i * 16 + l15;
      h8 lo = *(const h8*)&As[buf][m * LDSS + hh * 8];       // K = hh*8 .. +7
      h8 hi = *(const h8*)&As[buf][m * LDSS + 16 + hh * 8];  // K = 16+hh*8 .. +7
      af[i] = __builtin_shufflevector(lo, hi, 0,1,2,3,4,5,6,7,8,9,10,11,12,13,14,15);
    }
#pragma unroll
    for (int j = 0; j < 4; ++j) {
      int n = wn * 64 + j * 16 + l15;
      h8 lo = *(const h8*)&Bs[buf][n * LDSS + hh * 8];       // K = hh*8 .. +7
      h8 hi = *(const h8*)&Bs[buf][n * LDSS + 16 + hh * 8];  // K = 16+hh*8 .. +7
      bf[j] = __builtin_shufflevector(lo, hi, 0,1,2,3,4,5,6,7,8,9,10,11,12,13,14,15);
    }
#pragma unroll
    for (int i = 0; i < 2; ++i)
#pragma unroll
      for (int j = 0; j < 4; ++j)
        acc[i][j] = __builtin_amdgcn_wmma_f32_16x16x32_f16(
            false, af[i], false, bf[j], (short)0, acc[i][j], false, false);
  }

#pragma unroll
  for (int j = 0; j < 4; ++j) {
    int col = colBase + wn * 64 + j * 16 + l15;
    if (col < N) {
#pragma unroll
      for (int i = 0; i < 2; ++i) {
#pragma unroll
        for (int r = 0; r < 8; ++r) {
          int row = rowBase + wm * 32 + i * 16 + r + 8 * hh;
          C[(size_t)row * ldc + col] = acc[i][j][r];
        }
      }
    }
  }
}

// --------------------- f32 -> f16 convert (contiguous) ----------------------
__global__ void __launch_bounds__(256)
cvt_f16_kernel(const float* __restrict__ in, _Float16* __restrict__ out)
{
  int idx = blockIdx.x * 256 + threadIdx.x;
  out[idx] = (_Float16)in[idx];
}

// ----------------- tiled transpose + convert: [K][N] -> [N][K] --------------
__global__ void __launch_bounds__(256)
transpose_cvt_kernel(const float* __restrict__ in, _Float16* __restrict__ out,
                     int K, int N)
{
  __shared__ float tile[32][33];
  int k0 = blockIdx.x * 32, n0 = blockIdx.y * 32;
  int tx = threadIdx.x & 31, ty = threadIdx.x >> 5;   // 32 x 8
#pragma unroll
  for (int r = 0; r < 32; r += 8)
    tile[ty + r][tx] = in[(size_t)(k0 + ty + r) * N + n0 + tx];
  __syncthreads();
#pragma unroll
  for (int r = 0; r < 32; r += 8)
    out[(size_t)(n0 + ty + r) * K + k0 + tx] = (_Float16)tile[tx][ty + r];
}

// ------------------- depthwise causal conv (k=4) + SiLU ---------------------
__global__ void __launch_bounds__(256)
conv_silu_kernel(const float* __restrict__ xr, const float* __restrict__ Wc,
                 const float* __restrict__ bc, float* __restrict__ xs,
                 _Float16* __restrict__ xs16)
{
  int idx = blockIdx.x * 256 + threadIdx.x;   // over 4096*2048
  int c   = idx & 2047;
  int row = idx >> 11;
  int b   = row >> 11;
  int tl  = row & 2047;
  float acc = bc[c];
#pragma unroll
  for (int j = 0; j < 4; ++j) {
    int ts = tl - 3 + j;
    if (ts >= 0)
      acc += Wc[c * 4 + j] * xr[((size_t)(b * 2048 + ts)) * 4096 + c];
  }
  float s = acc / (1.0f + __expf(-acc));
  xs[idx]   = s;
  xs16[idx] = (_Float16)s;
}

// ------------------------- bias + softplus (in place) -----------------------
__global__ void __launch_bounds__(256)
softplus_bias_kernel(float* __restrict__ delta, const float* __restrict__ b_dt)
{
  int idx = blockIdx.x * 256 + threadIdx.x;
  int c   = idx & 2047;
  float v = delta[idx] + b_dt[c];
  delta[idx] = (v > 0.f) ? (v + log1pf(__expf(-v))) : log1pf(__expf(v));
}

// ------------------------ chunked selective scan ----------------------------
// 16 chunks x 128 steps. Diagonal SSM: prod_t exp(dt_t*A) == exp(A * sum dt).
// passA: per-chunk local terminal state (h0 = 0) + sum(dt)    [256 blocks]
// combine: h_start(c+1) = exp(A*S_c)*h_start(c) + L_c          [65536 thr]
// passB: replay chunk with correct h_start, emit y              [256 blocks]
#define TCH 128
#define NCH 16

__global__ void __launch_bounds__(256)
scan_passA(const float* __restrict__ xs, const float* __restrict__ delta,
           const float* __restrict__ xdbl, const float* __restrict__ A_log,
           float* __restrict__ Lbuf, float* __restrict__ Sbuf)
{
  __shared__ float sB[8][16];
  __shared__ float sC[8][16];
  const int tid = threadIdx.x;
  const int blk = blockIdx.x;            // 256 = 2 batch * 8 grp * 16 chunk
  const int cch = blk & 15;
  const int grp = (blk >> 4) & 7;
  const int b0  = blk >> 7;
  const int d   = grp * 256 + tid;
  const int g   = b0 * 2048 + d;

  float Av[16], h[16];
#pragma unroll
  for (int n = 0; n < 16; ++n) {
    Av[n] = -__expf(A_log[d * 16 + n]);
    h[n]  = 0.f;
  }
  float Ssum = 0.f;
  const int rb0 = b0 * 2048 + cch * TCH;

  for (int t0 = 0; t0 < TCH; t0 += 8) {
    {
      int step = tid >> 5;
      int idx  = tid & 31;
      float v  = xdbl[(size_t)(rb0 + t0 + step) * 96 + 64 + idx];
      if (idx < 16) sB[step][idx] = v; else sC[step][idx - 16] = v;
    }
    __syncthreads();
#pragma unroll
    for (int s = 0; s < 8; ++s) {
      size_t row = (size_t)(rb0 + t0 + s);
      float dt  = delta[row * 2048 + d];
      float xv  = xs[row * 2048 + d];
      float dtx = dt * xv;
      Ssum += dt;
#pragma unroll
      for (int n = 0; n < 16; ++n)
        h[n] = __expf(dt * Av[n]) * h[n] + dtx * sB[s][n];
    }
    __syncthreads();
  }
#pragma unroll
  for (int n = 0; n < 16; ++n)
    Lbuf[(size_t)g * 256 + cch * 16 + n] = h[n];
  Sbuf[g * 16 + cch] = Ssum;
}

__global__ void __launch_bounds__(256)
scan_combine(const float* __restrict__ A_log, const float* __restrict__ Sbuf,
             const float* __restrict__ Lbuf, float* __restrict__ Hstart)
{
  int idx = blockIdx.x * 256 + threadIdx.x;  // 65536 = 4096 ch * 16 states
  int g   = idx >> 4;
  int n   = idx & 15;
  int d   = g & 2047;
  float Avn = -__expf(A_log[d * 16 + n]);
  float h = 0.f;
  for (int c = 0; c < NCH; ++c) {
    Hstart[(size_t)g * 256 + c * 16 + n] = h;
    h = __expf(Avn * Sbuf[g * 16 + c]) * h + Lbuf[(size_t)g * 256 + c * 16 + n];
  }
}

__global__ void __launch_bounds__(256)
scan_passB(const float* __restrict__ xs, const float* __restrict__ delta,
           const float* __restrict__ xdbl, const float* __restrict__ A_log,
           const float* __restrict__ D, const float* __restrict__ Hstart,
           float* __restrict__ ys)
{
  __shared__ float sB[8][16];
  __shared__ float sC[8][16];
  const int tid = threadIdx.x;
  const int blk = blockIdx.x;
  const int cch = blk & 15;
  const int grp = (blk >> 4) & 7;
  const int b0  = blk >> 7;
  const int d   = grp * 256 + tid;
  const int g   = b0 * 2048 + d;

  float Av[16], h[16];
#pragma unroll
  for (int n = 0; n < 16; ++n) {
    Av[n] = -__expf(A_log[d * 16 + n]);
    h[n]  = Hstart[(size_t)g * 256 + cch * 16 + n];
  }
  const float Dv  = D[d];
  const int   rb0 = b0 * 2048 + cch * TCH;

  for (int t0 = 0; t0 < TCH; t0 += 8) {
    {
      int step = tid >> 5;
      int idx  = tid & 31;
      float v  = xdbl[(size_t)(rb0 + t0 + step) * 96 + 64 + idx];
      if (idx < 16) sB[step][idx] = v; else sC[step][idx - 16] = v;
    }
    __syncthreads();
#pragma unroll
    for (int s = 0; s < 8; ++s) {
      size_t row = (size_t)(rb0 + t0 + s);
      float dt  = delta[row * 2048 + d];
      float xv  = xs[row * 2048 + d];
      float dtx = dt * xv;
      float y   = 0.f;
#pragma unroll
      for (int n = 0; n < 16; ++n) {
        h[n] = __expf(dt * Av[n]) * h[n] + dtx * sB[s][n];
        y   += h[n] * sC[s][n];
      }
      ys[row * 2048 + d] = y + Dv * xv;
    }
    __syncthreads();
  }
}

// --------------------- gate: yact16 = f16(ys * silu(res)) -------------------
__global__ void __launch_bounds__(256)
gate_kernel(const float* __restrict__ ys, const float* __restrict__ xr,
            _Float16* __restrict__ yact16)
{
  int idx = blockIdx.x * 256 + threadIdx.x;   // over 4096*2048
  int c   = idx & 2047;
  int row = idx >> 11;
  float r = xr[(size_t)row * 4096 + 2048 + c];
  float s = r / (1.0f + __expf(-r));
  yact16[idx] = (_Float16)(ys[idx] * s);
}

// ------------------------------- launcher -----------------------------------
extern "C" void kernel_launch(void* const* d_in, const int* in_sizes, int n_in,
                              void* d_out, int out_size, void* d_ws, size_t ws_size,
                              hipStream_t stream)
{
  const float* x      = (const float*)d_in[0];   // (2,2048,1024)
  const float* W_in   = (const float*)d_in[1];   // (1024,4096)
  const float* W_conv = (const float*)d_in[2];   // (2048,1,4)
  const float* b_conv = (const float*)d_in[3];   // (2048,)
  const float* W_x    = (const float*)d_in[4];   // (2048,96)
  const float* W_dt   = (const float*)d_in[5];   // (64,2048)
  const float* b_dt   = (const float*)d_in[6];   // (2048,)
  const float* A_log  = (const float*)d_in[7];   // (2048,16)
  const float* Dp     = (const float*)d_in[8];   // (2048,)
  const float* W_out  = (const float*)d_in[9];   // (2048,1024)
  float* out = (float*)d_out;                    // (2,2048,1024)

  // ---- workspace carve-up (byte offsets, 256B aligned) ----
  char* ws = (char*)d_ws;
  size_t off = 0;
  auto alloc = [&](size_t bytes) { size_t o = off; off = (off + bytes + 255) & ~(size_t)255; return o; };

  float*    xr     = (float*)   (ws + alloc((size_t)4096*4096*4));
  float*    xs     = (float*)   (ws + alloc((size_t)4096*2048*4));
  float*    xdbl   = (float*)   (ws + alloc((size_t)4096*96*4));
  float*    delta  = (float*)   (ws + alloc((size_t)4096*2048*4));
  float*    ysb    = (float*)   (ws + alloc((size_t)4096*2048*4));
  float*    Lbuf   = (float*)   (ws + alloc((size_t)4096*NCH*16*4));
  float*    Hstart = (float*)   (ws + alloc((size_t)4096*NCH*16*4));
  float*    Sbuf   = (float*)   (ws + alloc((size_t)4096*NCH*4));
  _Float16* x16    = (_Float16*)(ws + alloc((size_t)4096*1024*2));
  _Float16* Win16T = (_Float16*)(ws + alloc((size_t)4096*1024*2));
  _Float16* xs16   = (_Float16*)(ws + alloc((size_t)4096*2048*2));
  _Float16* Wx16T  = (_Float16*)(ws + alloc((size_t)96*2048*2));
  _Float16* xdbl16 = (_Float16*)(ws + alloc((size_t)4096*96*2));
  _Float16* Wdt16T = (_Float16*)(ws + alloc((size_t)2048*64*2));
  _Float16* Wout16T= (_Float16*)(ws + alloc((size_t)1024*2048*2));
  _Float16* yact16 = (_Float16*)delta;   // delta dead after scan; reuse storage

  // ---- operand preparation (f16 + weight transposes) ----
  cvt_f16_kernel<<<(4096*1024)/256, 256, 0, stream>>>(x, x16);
  { dim3 g(1024/32, 4096/32); transpose_cvt_kernel<<<g, 256, 0, stream>>>(W_in, Win16T, 1024, 4096); }
  { dim3 g(2048/32,   96/32); transpose_cvt_kernel<<<g, 256, 0, stream>>>(W_x,  Wx16T,  2048, 96);   }
  { dim3 g(  64/32, 2048/32); transpose_cvt_kernel<<<g, 256, 0, stream>>>(W_dt, Wdt16T, 64,   2048); }
  { dim3 g(2048/32, 1024/32); transpose_cvt_kernel<<<g, 256, 0, stream>>>(W_out,Wout16T,2048, 1024); }

  // 1) xr = x @ W_in   [4096 x 4096], K=1024
  { dim3 grid(4096/BN, 4096/BM);
    gemm_wmma_f16<<<grid, 256, 0, stream>>>(x16, 1024, Win16T, 1024, xr, 4096,
                                            4096, 4096, 1024); }
  // 2) xs = silu(conv(xin)+b)  (also xs16)
  conv_silu_kernel<<<(4096*2048)/256, 256, 0, stream>>>(xr, W_conv, b_conv, xs, xs16);

  // 3) xdbl = xs @ W_x  [4096 x 96], K=2048
  { dim3 grid((96+BN-1)/BN, 4096/BM);
    gemm_wmma_f16<<<grid, 256, 0, stream>>>(xs16, 2048, Wx16T, 2048, xdbl, 96,
                                            4096, 96, 2048); }
  cvt_f16_kernel<<<(4096*96)/256, 256, 0, stream>>>(xdbl, xdbl16);

  // 4) delta = softplus(xdbl[:, :64] @ W_dt + b_dt)  [4096 x 2048], K=64
  { dim3 grid(2048/BN, 4096/BM);
    gemm_wmma_f16<<<grid, 256, 0, stream>>>(xdbl16, 96, Wdt16T, 64, delta, 2048,
                                            4096, 2048, 64); }
  softplus_bias_kernel<<<(4096*2048)/256, 256, 0, stream>>>(delta, b_dt);

  // 5) chunked selective scan -> ys
  scan_passA<<<256, 256, 0, stream>>>(xs, delta, xdbl, A_log, Lbuf, Sbuf);
  scan_combine<<<(4096*16)/256, 256, 0, stream>>>(A_log, Sbuf, Lbuf, Hstart);
  scan_passB<<<256, 256, 0, stream>>>(xs, delta, xdbl, A_log, Dp, Hstart, ysb);

  // 6) out = (ys * silu(res)) @ W_out   [4096 x 1024], K=2048
  gate_kernel<<<(4096*2048)/256, 256, 0, stream>>>(ysb, xr, yact16);
  { dim3 grid(1024/BN, 4096/BM);
    gemm_wmma_f16<<<grid, 256, 0, stream>>>(yact16, 2048, Wout16T, 2048, out, 1024,
                                            4096, 1024, 2048); }
}
